// RelativePositionAttention_6416681140952
// MI455X (gfx1250) — matrix operand order repaired and checked by reading, not compiled, not run on hardware
//
#include <hip/hip_runtime.h>
#include <hip/hip_bf16.h>

// ---------------------------------------------------------------------------
// RelativePositionAttention for MI455X (gfx1250), wave32 + WMMA f16->f32.
// B=4, N=2304 (48x48), C=384, heads=8, hd=48 (padded to 64), in=448.
// All WMMA operand fetches are 2x b128 loads (LDS, but Q from global); attention uses an LDS
// decay table + double-buffered K/V tiles with register prefetch.
// ---------------------------------------------------------------------------

typedef __attribute__((ext_vector_type(16))) _Float16 v16h;
typedef __attribute__((ext_vector_type(8)))  float    v8f;

#define NB   4
#define NH   8
#define NN   2304
#define NC   384
#define PD   64
#define KIN  448
#define HD   48
#define HDP  64
#define SGRID 48         // 48x48 spatial grid
#define NKT  (NN / 64)   // 36 key tiles
#define SCALE 0.14433756729740643f  // 48^-0.5
#define INV47 0.021276595744680851f // 1/47 (linspace(0,1,48) step)
#define DT_W 96          // decay table row stride (dx rows x dy cols)
#define DT_N (95 * DT_W) // 9120 entries
#define DT_OFF (47 * DT_W + 47)

union H16 { uint4 u[2]; v16h v; _Float16 h[16]; };

// A (16x32 f16) fragment: per lane 8 contiguous + 8 contiguous (ISA 7.12.2).
__device__ inline v16h load_A(const _Float16* s, int ld, int koff) {
  int lane = threadIdx.x & 31;
  const _Float16* p = s + (lane & 15) * ld + koff + ((lane >> 4) << 3);
  H16 r;
  r.u[0] = *(const uint4*)(p);
  r.u[1] = *(const uint4*)(p + 16);
  return r.v;
}
// B (32x16 f16) fragment from COLUMN-MAJOR storage s[n][k] (row stride ld).
__device__ inline v16h load_Bcm(const _Float16* s, int ld, int koff) {
  int lane = threadIdx.x & 31;
  const _Float16* p = s + (lane & 15) * ld + koff + ((lane >> 4) << 3);
  H16 r;
  r.u[0] = *(const uint4*)(p);
  r.u[1] = *(const uint4*)(p + 16);
  return r.v;
}

__device__ inline float rowred_max(float v) {
#pragma unroll
  for (int m = 1; m <= 8; m <<= 1) v = fmaxf(v, __shfl_xor(v, m, 32));
  return v;
}
__device__ inline float rowred_sum(float v) {
#pragma unroll
  for (int m = 1; m <= 8; m <<= 1) v += __shfl_xor(v, m, 32);
  return v;
}

// --- workspace zero (padded head-dim rows/cols must be 0) -------------------
__global__ void zero_kernel(uint4* p, size_t n) {
  size_t i = (size_t)blockIdx.x * blockDim.x + threadIdx.x;
  uint4 z; z.x = 0; z.y = 0; z.z = 0; z.w = 0;
  for (; i < n; i += (size_t)gridDim.x * blockDim.x) p[i] = z;
}

// --- kernel 1: fused QKV projection ----------------------------------------
__global__ __launch_bounds__(128) void qkv_kernel(
    const float* __restrict__ x, const float* __restrict__ pos,
    const float* __restrict__ Wq, const float* __restrict__ Wkv,
    _Float16* __restrict__ Q, _Float16* __restrict__ K, _Float16* __restrict__ V) {
  __shared__ __align__(16) _Float16 At[64][40];   // A row-major [row][k]
  __shared__ __align__(16) _Float16 Bl[64][40];   // B col-major [col][k]
  const int t = threadIdx.x, wv = t >> 5, lane = t & 31;
  const int col = lane & 15;
  const int rowbase = blockIdx.y * 64;
  const int colbase = blockIdx.x * 64;
  v8f acc[4] = {};
  for (int kb = 0; kb < KIN; kb += 32) {
    __syncthreads();
    {
      int r = t >> 1, c0 = (t & 1) * 16;
      int gr = rowbase + r, b = gr / NN, n = gr % NN;
      int k0 = kb + c0;
      const float* g = (k0 < NC) ? (x + ((size_t)b * NN + n) * NC + k0)
                                 : (pos + (size_t)n * PD + (k0 - NC));
      H16 tmp;
#pragma unroll
      for (int i = 0; i < 16; ++i) tmp.h[i] = (_Float16)g[i];
      uint4* d = (uint4*)&At[r][c0];
      d[0] = tmp.u[0]; d[1] = tmp.u[1];
    }
    {
      int cl = t & 63, kh = (t >> 6) * 16;
      int gc = colbase + cl;
      const float* w; int str;
      if (gc < NC) { w = Wq + (size_t)(kb + kh) * NC + gc; str = NC; }
      else { w = Wkv + (size_t)(kb + kh) * (2 * NC) + (gc - NC); str = 2 * NC; }
      H16 tmp;
#pragma unroll
      for (int i = 0; i < 16; ++i) tmp.h[i] = (_Float16)w[(size_t)i * str];
      uint4* d = (uint4*)&Bl[cl][kh];
      d[0] = tmp.u[0]; d[1] = tmp.u[1];
    }
    __syncthreads();
    v16h a0 = load_A(&At[wv * 16][0], 40, 0);
#pragma unroll
    for (int j = 0; j < 4; ++j)
      acc[j] = __builtin_amdgcn_wmma_f32_16x16x32_f16(
          false, a0, false, load_Bcm(&Bl[j * 16][0], 40, 0), (short)0, acc[j], false, false);
  }
  const int rbase = (lane >> 4) << 3;
#pragma unroll
  for (int j = 0; j < 4; ++j)
#pragma unroll
    for (int i = 0; i < 8; ++i) {
      int gr = rowbase + wv * 16 + rbase + i;
      int b = gr / NN, n = gr % NN;
      int gc = colbase + j * 16 + col;
      _Float16 hv = (_Float16)acc[j][i];
      if (gc < NC) {
        int h = gc / HD, d = gc % HD;
        Q[(((size_t)b * NH + h) * NN + n) * HDP + d] = hv;
      } else if (gc < 2 * NC) {
        int c2 = gc - NC, h = c2 / HD, d = c2 % HD;
        K[(((size_t)b * NH + h) * NN + n) * HDP + d] = hv;
      } else {
        int c2 = gc - 2 * NC, h = c2 / HD, d = c2 % HD;
        V[(((size_t)b * NH + h) * HDP + d) * NN + n] = hv;  // transposed
      }
    }
}

// --- kernel 2: streaming attention with LDS decay table ---------------------
// out = (sum_m exp(s-m)*Dtab[dq-dk]*v) / (sum_m exp(s-m))
__global__ __launch_bounds__(128) void attn_kernel(
    const _Float16* __restrict__ Q, const _Float16* __restrict__ K,
    const _Float16* __restrict__ V, const float* __restrict__ alpha_p,
    _Float16* __restrict__ O) {
  __shared__ __align__(16) _Float16 Kt[2][64][72];   // [buf][key][d]
  __shared__ __align__(16) _Float16 Vt[2][64][72];   // [buf][d][key]
  __shared__ __align__(16) _Float16 Pt[4][16][72];   // per-wave P' row-major
  __shared__ float Dtab[DT_N];                       // exp(alpha*dist(dx,dy))
  const int t = threadIdx.x, wv = t >> 5, lane = t & 31;
  const int col = lane & 15;
  const int rbase = (lane >> 4) << 3;
  const int bh = blockIdx.y;
  const int qbase = blockIdx.x * 64;
  const float alpha = alpha_p[0];

  // decay table: depends only on (dx,dy); computed once per block
  for (int idx = t; idx < DT_N; idx += 128) {
    int dx = idx / DT_W - 47, dy = idx % DT_W - 47;
    Dtab[idx] = __expf(alpha * sqrtf((float)(dx * dx + dy * dy)) * INV47);
  }

  const _Float16* Qp = Q + ((size_t)bh * NN + qbase + wv * 16) * HDP;
  v16h aq0 = load_A(Qp, HDP, 0);
  v16h aq1 = load_A(Qp, HDP, 32);

  int qlin[8];
  float mrow[8], lrow[8];
#pragma unroll
  for (int i = 0; i < 8; ++i) {
    int n = qbase + wv * 16 + rbase + i;
    qlin[i] = (n / SGRID) * DT_W + (n % SGRID) + DT_OFF;
    mrow[i] = -3.0e38f; lrow[i] = 0.0f;
  }
  // per-tile key coordinates, updated incrementally (+64 = +1 row +16 cols)
  int kx[4], ky[4];
#pragma unroll
  for (int j = 0; j < 4; ++j) { int n = j * 16 + col; kx[j] = n / SGRID; ky[j] = n % SGRID; }

  v8f acc[4] = {};
  const int sr = t >> 1, sh = (t & 1) * 32;   // staging row/half

  // prefetch tile 0 into registers
  uint4 rk[4], rv[4];
  {
    const uint4* gk = (const uint4*)(K + ((size_t)bh * NN + sr) * HDP + sh);
    const uint4* gv = (const uint4*)(V + ((size_t)bh * HDP + sr) * NN + sh);
#pragma unroll
    for (int i = 0; i < 4; ++i) { rk[i] = gk[i]; rv[i] = gv[i]; }
  }

  int buf = 0;
  for (int kt = 0; kt < NKT; ++kt) {
    { // commit prefetched tile to LDS
      uint4* sk = (uint4*)(&Kt[buf][sr][sh]);
      uint4* sv = (uint4*)(&Vt[buf][sr][sh]);
#pragma unroll
      for (int i = 0; i < 4; ++i) { sk[i] = rk[i]; sv[i] = rv[i]; }
    }
    __syncthreads();
    if (kt + 1 < NKT) { // prefetch next tile (overlaps with compute below)
      const uint4* gk = (const uint4*)(K + ((size_t)bh * NN + (kt + 1) * 64 + sr) * HDP + sh);
      const uint4* gv = (const uint4*)(V + ((size_t)bh * HDP + sr) * NN + (kt + 1) * 64 + sh);
#pragma unroll
      for (int i = 0; i < 4; ++i) { rk[i] = gk[i]; rv[i] = gv[i]; }
    }

    // S = Q * K^T (16x64): 4 col blocks x 2 K-steps
    v8f sb[4];
#pragma unroll
    for (int j = 0; j < 4; ++j) {
      v8f z = {};
      v8f s0 = __builtin_amdgcn_wmma_f32_16x16x32_f16(
          false, aq0, false, load_Bcm(&Kt[buf][j * 16][0], 72, 0), (short)0, z, false, false);
      sb[j] = __builtin_amdgcn_wmma_f32_16x16x32_f16(
          false, aq1, false, load_Bcm(&Kt[buf][j * 16][0], 72, 32), (short)0, s0, false, false);
    }

    // online softmax over the 64-key tile
    float mloc[8];
#pragma unroll
    for (int i = 0; i < 8; ++i) {
      float mm = -3.0e38f;
#pragma unroll
      for (int j = 0; j < 4; ++j) { sb[j][i] = sb[j][i] * SCALE; mm = fmaxf(mm, sb[j][i]); }
      mloc[i] = rowred_max(mm);
    }
    int klin[4];
#pragma unroll
    for (int j = 0; j < 4; ++j) klin[j] = kx[j] * DT_W + ky[j];
#pragma unroll
    for (int i = 0; i < 8; ++i) {
      float mn = fmaxf(mrow[i], mloc[i]);
      float rex = __expf(mrow[i] - mn);
      mrow[i] = mn;
      float ps = 0.0f;
#pragma unroll
      for (int j = 0; j < 4; ++j) {
        float p = __expf(sb[j][i] - mn);
        ps += p;
        float dcy = Dtab[qlin[i] - klin[j]];    // exp(alpha*dist) via table
        Pt[wv][rbase + i][j * 16 + col] = (_Float16)(p * dcy);
      }
      lrow[i] = lrow[i] * rex + ps;             // lane-partial; reduced at end
#pragma unroll
      for (int j = 0; j < 4; ++j) acc[j][i] = acc[j][i] * rex;
    }

    // acc += P' * V  (wave-private LDS round-trip; DS ops in-order per wave)
    v16h pa0 = load_A(&Pt[wv][0][0], 72, 0);
    v16h pa1 = load_A(&Pt[wv][0][0], 72, 32);
#pragma unroll
    for (int j = 0; j < 4; ++j) {
      acc[j] = __builtin_amdgcn_wmma_f32_16x16x32_f16(
          false, pa0, false, load_Bcm(&Vt[buf][j * 16][0], 72, 0), (short)0, acc[j], false, false);
      acc[j] = __builtin_amdgcn_wmma_f32_16x16x32_f16(
          false, pa1, false, load_Bcm(&Vt[buf][j * 16][0], 72, 32), (short)0, acc[j], false, false);
    }

    // advance key coordinates by 64 = (+1, +16) with carry
#pragma unroll
    for (int j = 0; j < 4; ++j) {
      ky[j] += 16; kx[j] += 1;
      if (ky[j] >= SGRID) { ky[j] -= SGRID; kx[j] += 1; }
    }
    buf ^= 1;
  }

  // finalize: full-row denominator via deferred cross-lane reduction
#pragma unroll
  for (int i = 0; i < 8; ++i) lrow[i] = rowred_sum(lrow[i]);
#pragma unroll
  for (int j = 0; j < 4; ++j)
#pragma unroll
    for (int i = 0; i < 8; ++i) {
      int n = qbase + wv * 16 + rbase + i;
      float o = acc[j][i] / lrow[i];
      O[((size_t)bh * NN + n) * HDP + j * 16 + col] = (_Float16)o;
    }
}

// --- kernel 3: output projection [9216x384] @ [384x384] + bias -> f32 -------
__global__ __launch_bounds__(128) void proj_kernel(
    const _Float16* __restrict__ O, const float* __restrict__ Wp,
    const float* __restrict__ bias, float* __restrict__ out) {
  __shared__ __align__(16) _Float16 At[64][40];
  __shared__ __align__(16) _Float16 Bl[64][40];
  const int t = threadIdx.x, wv = t >> 5, lane = t & 31;
  const int col = lane & 15;
  const int rowbase = blockIdx.y * 64;
  const int colbase = blockIdx.x * 64;
  v8f acc[4] = {};
  for (int kb = 0; kb < NC; kb += 32) {
    __syncthreads();
    {
      int r = t >> 1, c0 = (t & 1) * 16;
      int gr = rowbase + r, b = gr / NN, n = gr % NN;
      H16 tmp;
#pragma unroll
      for (int i = 0; i < 16; ++i) {
        int k = kb + c0 + i, h = k / HD, d = k % HD;
        tmp.h[i] = O[(((size_t)b * NH + h) * NN + n) * HDP + d];
      }
      uint4* dst = (uint4*)&At[r][c0];
      dst[0] = tmp.u[0]; dst[1] = tmp.u[1];
    }
    {
      int cl = t & 63, kh = (t >> 6) * 16;
      const float* w = Wp + (size_t)(kb + kh) * NC + colbase + cl;
      H16 tmp;
#pragma unroll
      for (int i = 0; i < 16; ++i) tmp.h[i] = (_Float16)w[(size_t)i * NC];
      uint4* d = (uint4*)&Bl[cl][kh];
      d[0] = tmp.u[0]; d[1] = tmp.u[1];
    }
    __syncthreads();
    v16h a0 = load_A(&At[wv * 16][0], 40, 0);
#pragma unroll
    for (int j = 0; j < 4; ++j)
      acc[j] = __builtin_amdgcn_wmma_f32_16x16x32_f16(
          false, a0, false, load_Bcm(&Bl[j * 16][0], 40, 0), (short)0, acc[j], false, false);
  }
  const int rbase = (lane >> 4) << 3;
#pragma unroll
  for (int j = 0; j < 4; ++j)
#pragma unroll
    for (int i = 0; i < 8; ++i) {
      int gr = rowbase + wv * 16 + rbase + i;
      int gc = colbase + j * 16 + col;
      out[(size_t)gr * NC + gc] = acc[j][i] + bias[gc];
    }
}

// ---------------------------------------------------------------------------
extern "C" void kernel_launch(void* const* d_in, const int* in_sizes, int n_in,
                              void* d_out, int out_size, void* d_ws, size_t ws_size,
                              hipStream_t stream) {
  const float* x    = (const float*)d_in[0];
  const float* pos  = (const float*)d_in[1];
  const float* alph = (const float*)d_in[2];
  const float* Wq   = (const float*)d_in[3];
  const float* Wkv  = (const float*)d_in[4];
  const float* Wp   = (const float*)d_in[5];
  const float* bp   = (const float*)d_in[6];
  float* out = (float*)d_out;

  const size_t per = (size_t)NB * NH * NN * HDP;
  _Float16* Q = (_Float16*)d_ws;
  _Float16* K = Q + per;
  _Float16* V = K + per;
  _Float16* O = V + per;

  zero_kernel<<<2048, 256, 0, stream>>>((uint4*)d_ws, (3 * per * sizeof(_Float16)) / 16);
  qkv_kernel<<<dim3((3 * NC) / 64, (NB * NN) / 64), 128, 0, stream>>>(x, pos, Wq, Wkv, Q, K, V);
  attn_kernel<<<dim3(NN / 64, NB * NH), 128, 0, stream>>>(Q, K, V, alph, O);
  proj_kernel<<<dim3(NC / 64, (NB * NN) / 64), 128, 0, stream>>>(O, Wp, bp, out);
}
